// GraphAttentionLayer_70763881169253
// MI455X (gfx1250) — matrix operand order repaired and checked by reading, not compiled, not run on hardware
//
// GraphAttentionLayer fused bf16-WMMA implementation for gfx1250 (MI455X).
// Pipeline: pack weights (bf16 B-frag swizzle) -> QKV GEMM (LDS-staged A frags)
//        -> flash attention (online softmax in exp2 domain, adjacency edge bias,
//           all matmuls on v_wmma_f32_16x16x32_bf16) -> Wo GEMM + residual + LN.
// Round 2: software-pipelined fragment loads (double-buffered B frags in GEMMs,
// ping-pong K frags + early V/adj issue in attention) to break the
// load->s_wait_loadcnt(0)->wmma serialization seen in round-1 disassembly.
#include <hip/hip_runtime.h>

#define D_MODEL 256
#define N_HEADS 4
#define SEQ     2048
#define BATCH   4

typedef __attribute__((ext_vector_type(16))) __bf16   v16bf;
typedef __attribute__((ext_vector_type(8)))  float    v8f;
typedef __attribute__((ext_vector_type(4)))  unsigned u32x4;

union FragU { u32x4 q[2]; v16bf v; };

#if __has_builtin(__builtin_amdgcn_exp2f)
#define EXP2F(x) __builtin_amdgcn_exp2f(x)
#else
#define EXP2F(x) exp2f(x)
#endif
#define LOG2E 1.44269504088896340736f

__device__ __forceinline__ unsigned short f2bf(float x) {
  unsigned u = __float_as_uint(x);
  u += 0x7FFFu + ((u >> 16) & 1u);           // round-to-nearest-even
  return (unsigned short)(u >> 16);
}
__device__ __forceinline__ unsigned pk2(float a, float b) {
  return (unsigned)f2bf(a) | ((unsigned)f2bf(b) << 16);
}
__device__ __forceinline__ v8f wmma_bf16(const FragU& a, const FragU& b, v8f c) {
  return __builtin_amdgcn_wmma_f32_16x16x32_bf16(false, a.v, false, b.v,
                                                 (short)0, c, false, false);
}

// ---------------------------------------------------------------------------
// Kernel 0: pack fp32 weights into bf16 B-fragment layout.
// PW[mat][ct][kt][lane][j] : packed pair W[kt*32 + (j/4)*16 + (lane/16)*8 + 2(j%4) + {0,1}][ct*16 + lane%16]
// ---------------------------------------------------------------------------
__global__ __launch_bounds__(256) void pack_w(
    const float* __restrict__ Wq, const float* __restrict__ Wk,
    const float* __restrict__ Wv, const float* __restrict__ Wo,
    unsigned* __restrict__ PW)
{
  const int idx  = blockIdx.x * 256 + threadIdx.x;   // 0..131071
  const int j    = idx & 7;
  const int lane = (idx >> 3) & 31;
  const int kt   = (idx >> 8) & 7;
  const int ct   = (idx >> 11) & 15;
  const int mat  = idx >> 15;
  const float* W = (mat == 0) ? Wq : (mat == 1) ? Wk : (mat == 2) ? Wv : Wo;
  const int n  = ct * 16 + (lane & 15);
  const int k0 = kt * 32 + (j >> 2) * 16 + (lane >> 4) * 8 + 2 * (j & 3);
  PW[idx] = pk2(W[k0 * D_MODEL + n], W[(k0 + 1) * D_MODEL + n]);
}

// ---------------------------------------------------------------------------
// Kernel 1: QKV projection. One block = 16 tokens; 8 waves x 6 output tiles.
// Q/K written row-major bf16 [b][h][n][64]; V written pre-swizzled for P*V.
// B fragments double-buffered so global loads overlap WMMA.
// ---------------------------------------------------------------------------
__global__ __launch_bounds__(256) void gat_qkv(
    const float* __restrict__ X,
    const float* __restrict__ bq, const float* __restrict__ bk,
    const float* __restrict__ bvv,
    const unsigned* __restrict__ PW,
    unsigned short* __restrict__ Qp, unsigned short* __restrict__ Kp,
    unsigned* __restrict__ VP)
{
  __shared__ unsigned ldsx[16][128];                 // 16 x 256 bf16 packed pairs
  const int tid = threadIdx.x;
  const int g0  = blockIdx.x * 16;
  for (int p = tid; p < 2048; p += 256) {
    const int row = p >> 7, c = p & 127;
    const float* s = X + (size_t)(g0 + row) * D_MODEL + 2 * c;
    ldsx[row][c] = pk2(s[0], s[1]);
  }
  __syncthreads();

  const int wave = tid >> 5, lane = tid & 31;
  const int lm = lane & 15, lh = lane >> 4;
  const int bb = g0 / SEQ, tloc = g0 % SEQ;

  for (int i = 0; i < 6; ++i) {
    const int t = wave * 6 + i;                      // 0..47
    const int mat = t >> 4, ct = t & 15;
    const unsigned* pw = PW + mat * 32768 + ct * 2048 + lane * 8;

    FragU bb2[2];
    bb2[0].q[0] = *(const u32x4*)pw;
    bb2[0].q[1] = *(const u32x4*)(pw + 4);
    v8f c = {};
    #pragma unroll
    for (int kt = 0; kt < 8; ++kt) {
      const int cur = kt & 1;
      if (kt < 7) {                                  // prefetch next B fragment
        const unsigned* pb = pw + (kt + 1) * 256;
        bb2[cur ^ 1].q[0] = *(const u32x4*)pb;
        bb2[cur ^ 1].q[1] = *(const u32x4*)(pb + 4);
      }
      FragU a;
      a.q[0] = *(const u32x4*)&ldsx[lm][kt * 16 + lh * 4];
      a.q[1] = *(const u32x4*)&ldsx[lm][kt * 16 + 8 + lh * 4];
      c = wmma_bf16(a, bb2[cur], c);
    }
    const float* bias = (mat == 0) ? bq : (mat == 1) ? bk : bvv;
    const float badd = bias[ct * 16 + lm];
    #pragma unroll
    for (int r = 0; r < 8; ++r) c[r] += badd;

    if (mat < 2) {
      const int col = ct * 16 + lm;
      const int h = col >> 6, dl = col & 63;
      unsigned short* dst = (mat == 0) ? Qp : Kp;
      #pragma unroll
      for (int r = 0; r < 8; ++r) {
        const int token = tloc + lh * 8 + r;
        dst[((size_t)(bb * N_HEADS + h) * SEQ + token) * 64 + dl] = f2bf(c[r]);
      }
    } else {
      const int h = ct >> 2, cb = ct & 3;
      const int kt2 = tloc >> 5, khK = (tloc >> 4) & 1;
      const size_t idx =
          (((((size_t)(bb * N_HEADS + h) * 64 + kt2) * 2 + khK) * 4 + cb) * 16 + lm) * 8
          + lh * 4;
      u32x4 pkv;
      pkv.x = pk2(c[0], c[1]);
      pkv.y = pk2(c[2], c[3]);
      pkv.z = pk2(c[4], c[5]);
      pkv.w = pk2(c[6], c[7]);
      *(u32x4*)&VP[idx] = pkv;
    }
  }
}

// ---------------------------------------------------------------------------
// Kernel 2: fused flash attention with adjacency edge bias.
// One wave per (b, h, 16-query block); 4 waves of a block share (b,h) with
// consecutive query blocks (K/V locality). Loop over 64 key-blocks of 32.
// K fragments ping-pong across iterations; V/adjacency issued at loop top so
// softmax VALU overlays the memory latency. Softmax runs in exp2 domain.
// ---------------------------------------------------------------------------
__global__ __launch_bounds__(128) void gat_attn(
    const float* __restrict__ adj,
    const float* __restrict__ We, const float* __restrict__ be,
    const unsigned* __restrict__ Qu, const unsigned* __restrict__ Ku,
    const unsigned* __restrict__ VP,
    unsigned short* __restrict__ AO)
{
  __shared__ unsigned pbuf[4][16][16];               // per-wave P tile (16x32 bf16)
  const int tid = threadIdx.x;
  const int wave = tid >> 5, lane = tid & 31;
  const int lm = lane & 15, lh = lane >> 4;
  const int wg = blockIdx.x * 4 + wave;              // 0..2047
  const int qb = wg & 127;
  const int h  = (wg >> 7) & 3;
  const int bb = wg >> 9;
  const int q0 = qb * 16;
  const float cw = We[h] * LOG2E;                    // edge-bias weight (log2 domain)
  const float cb0 = be[h] * LOG2E;
  const float cs = 0.125f * LOG2E;                   // 1/sqrt(64) * log2(e)

  // Q A-fragments for full d=64 (two K=32 steps), loaded once.
  const unsigned* qr = Qu + (size_t)((bb * N_HEADS + h) * SEQ + q0 + lm) * 32;
  FragU aq0, aq1;
  aq0.q[0] = *(const u32x4*)(qr + lh * 4);
  aq0.q[1] = *(const u32x4*)(qr + 8 + lh * 4);
  aq1.q[0] = *(const u32x4*)(qr + 16 + lh * 4);
  aq1.q[1] = *(const u32x4*)(qr + 24 + lh * 4);

  v8f O[4] = {};
  v8f mrun, lrun;
  #pragma unroll
  for (int r = 0; r < 8; ++r) { mrun[r] = -3.0e38f; lrun[r] = 0.0f; }

  const size_t   adj_row0 = ((size_t)bb * SEQ + q0) * SEQ;
  const unsigned kvbase   = (bb * N_HEADS + h) * SEQ;
  const unsigned vpbase   = (bb * N_HEADS + h) * 64;

  auto loadK = [&](FragU* dst, int kt_) {
    #pragma unroll
    for (int nt = 0; nt < 2; ++nt) {
      const int key = kt_ * 32 + nt * 16 + lm;
      const unsigned* kr = Ku + (size_t)(kvbase + key) * 32;
      dst[nt * 2 + 0].q[0] = *(const u32x4*)(kr + lh * 4);
      dst[nt * 2 + 0].q[1] = *(const u32x4*)(kr + 8 + lh * 4);
      dst[nt * 2 + 1].q[0] = *(const u32x4*)(kr + 16 + lh * 4);
      dst[nt * 2 + 1].q[1] = *(const u32x4*)(kr + 24 + lh * 4);
    }
  };

  FragU kbuf[2][4];
  loadK(kbuf[0], 0);

  #pragma unroll 2
  for (int kt = 0; kt < 64; ++kt) {
    const int cur = kt & 1;

    // Issue V fragments + adjacency for this block up front (consumed late).
    FragU vf[4];
    const unsigned* vb = VP + (size_t)(((vpbase + kt) * 2 + lh) * 4) * 128;
    #pragma unroll
    for (int cbi = 0; cbi < 4; ++cbi) {
      const unsigned* p = vb + cbi * 128 + lm * 8;
      vf[cbi].q[0] = *(const u32x4*)p;
      vf[cbi].q[1] = *(const u32x4*)(p + 4);
    }
    float av0[8], av1[8];
    #pragma unroll
    for (int r = 0; r < 8; ++r) {
      const size_t rowo = adj_row0 + (size_t)(lh * 8 + r) * SEQ + kt * 32 + lm;
      av0[r] = adj[rowo];
      av1[r] = adj[rowo + 16];
    }
    // Prefetch K fragments for the next block (issued last -> waits overlap).
    loadK(kbuf[cur ^ 1], (kt + 1 < 64) ? kt + 1 : 63);
    if (kt + 1 < 64)
      __builtin_prefetch(&adj[adj_row0 + (size_t)(lh * 8) * SEQ + (kt + 1) * 32 + lm], 0, 1);

    // Scores (already resident K fragments from previous iteration).
    v8f s0 = {}, s1 = {};
    s0 = wmma_bf16(aq0, kbuf[cur][0], s0);
    s0 = wmma_bf16(aq1, kbuf[cur][1], s0);
    s1 = wmma_bf16(aq0, kbuf[cur][2], s1);
    s1 = wmma_bf16(aq1, kbuf[cur][3], s1);
    #pragma unroll
    for (int r = 0; r < 8; ++r) {
      s0[r] = s0[r] * cs + av0[r] * cw + cb0;        // log2-domain scores
      s1[r] = s1[r] * cs + av1[r] * cw + cb0;
    }

    // Row-wise max across the 16 lanes holding a row's key columns.
    v8f tmax;
    #pragma unroll
    for (int r = 0; r < 8; ++r) tmax[r] = fmaxf(s0[r], s1[r]);
    #pragma unroll
    for (int off = 1; off < 16; off <<= 1) {
      #pragma unroll
      for (int r = 0; r < 8; ++r)
        tmax[r] = fmaxf(tmax[r], __shfl_xor(tmax[r], off, 32));
    }
    v8f alpha;
    #pragma unroll
    for (int r = 0; r < 8; ++r) {
      const float mnew = fmaxf(mrun[r], tmax[r]);
      alpha[r] = EXP2F(mrun[r] - mnew);
      mrun[r]  = mnew;
      s0[r] = EXP2F(s0[r] - mnew);
      s1[r] = EXP2F(s1[r] - mnew);
    }
    v8f tsum;
    #pragma unroll
    for (int r = 0; r < 8; ++r) tsum[r] = s0[r] + s1[r];
    #pragma unroll
    for (int off = 1; off < 16; off <<= 1) {
      #pragma unroll
      for (int r = 0; r < 8; ++r) tsum[r] += __shfl_xor(tsum[r], off, 32);
    }
    #pragma unroll
    for (int r = 0; r < 8; ++r) {
      lrun[r] = lrun[r] * alpha[r] + tsum[r];
      O[0][r] *= alpha[r]; O[1][r] *= alpha[r];
      O[2][r] *= alpha[r]; O[3][r] *= alpha[r];
    }

    // Transpose P (C layout -> A layout) through LDS as bf16.
    unsigned short* pp = (unsigned short*)pbuf[wave];
    #pragma unroll
    for (int r = 0; r < 8; ++r) {
      pp[(lh * 8 + r) * 32 + lm]      = f2bf(s0[r]);
      pp[(lh * 8 + r) * 32 + 16 + lm] = f2bf(s1[r]);
    }
    asm volatile("s_wait_dscnt 0" ::: "memory");      // same-wave DS RAW fence
    FragU ap;
    const unsigned* pr = &pbuf[wave][lm][0];
    ap.q[0] = *(const u32x4*)(pr + lh * 8);
    ap.q[1] = *(const u32x4*)(pr + lh * 8 + 4);

    // O += P * V (V fragments were issued at the top of the iteration).
    #pragma unroll
    for (int cbi = 0; cbi < 4; ++cbi)
      O[cbi] = wmma_bf16(ap, vf[cbi], O[cbi]);
  }

  // Normalize and store pre-Wo activations (bf16 row-major [b][n][256]).
  #pragma unroll
  for (int r = 0; r < 8; ++r) {
    const float inv = 1.0f / lrun[r];
    const int token = q0 + lh * 8 + r;
    const size_t rowo = ((size_t)bb * SEQ + token) * D_MODEL + h * 64 + lm;
    AO[rowo +  0] = f2bf(O[0][r] * inv);
    AO[rowo + 16] = f2bf(O[1][r] * inv);
    AO[rowo + 32] = f2bf(O[2][r] * inv);
    AO[rowo + 48] = f2bf(O[3][r] * inv);
  }
}

// ---------------------------------------------------------------------------
// Kernel 3: output projection (WMMA) + bias + residual + LayerNorm.
// Both A and B fragments double-buffered across K-steps.
// ---------------------------------------------------------------------------
__global__ __launch_bounds__(256) void gat_out_ln(
    const float* __restrict__ X, const unsigned* __restrict__ AOu,
    const unsigned* __restrict__ PWo, const float* __restrict__ bo,
    const float* __restrict__ gamma, const float* __restrict__ beta,
    float* __restrict__ out)
{
  __shared__ float Y[16][D_MODEL];
  __shared__ float red[2][16][16];
  __shared__ float stats[2][16];
  const int tid = threadIdx.x;
  const int wave = tid >> 5, lane = tid & 31;
  const int lm = lane & 15, lh = lane >> 4;
  const int g0 = blockIdx.x * 16;
  const unsigned* ar = AOu + (size_t)(g0 + lm) * 128;

  for (int i = 0; i < 2; ++i) {
    const int ct = wave * 2 + i;
    const unsigned* pw = PWo + ct * 2048 + lane * 8;

    FragU ab[2], bb2[2];
    ab[0].q[0]  = *(const u32x4*)(ar + lh * 4);
    ab[0].q[1]  = *(const u32x4*)(ar + 8 + lh * 4);
    bb2[0].q[0] = *(const u32x4*)pw;
    bb2[0].q[1] = *(const u32x4*)(pw + 4);
    v8f c = {};
    #pragma unroll
    for (int kt = 0; kt < 8; ++kt) {
      const int cur = kt & 1;
      if (kt < 7) {
        const unsigned* pa = ar + (kt + 1) * 16 + lh * 4;
        const unsigned* pb = pw + (kt + 1) * 256;
        ab[cur ^ 1].q[0]  = *(const u32x4*)pa;
        ab[cur ^ 1].q[1]  = *(const u32x4*)(pa + 8);
        bb2[cur ^ 1].q[0] = *(const u32x4*)pb;
        bb2[cur ^ 1].q[1] = *(const u32x4*)(pb + 4);
      }
      c = wmma_bf16(ab[cur], bb2[cur], c);
    }
    const int col = ct * 16 + lm;
    const float badd = bo[col];
    #pragma unroll
    for (int r = 0; r < 8; ++r) {
      const int row = lh * 8 + r;
      Y[row][col] = c[r] + badd + X[(size_t)(g0 + row) * D_MODEL + col];
    }
  }
  __syncthreads();

  const int token = tid >> 4, seg = tid & 15;
  float s = 0.f, s2 = 0.f;
  #pragma unroll
  for (int c = 0; c < 16; ++c) {
    const float y = Y[token][seg * 16 + c];
    s += y; s2 += y * y;
  }
  red[0][token][seg] = s; red[1][token][seg] = s2;
  __syncthreads();
  if (tid < 16) {
    float a = 0.f, b = 0.f;
    #pragma unroll
    for (int i = 0; i < 16; ++i) { a += red[0][tid][i]; b += red[1][tid][i]; }
    const float mu  = a * (1.0f / 256.0f);
    const float var = b * (1.0f / 256.0f) - mu * mu;
    stats[0][tid] = mu;
    stats[1][tid] = rsqrtf(var + 1e-5f);
  }
  __syncthreads();
  const float mu = stats[0][token], rs = stats[1][token];
  for (int c = 0; c < 16; ++c) {
    const int col = seg * 16 + c;
    out[(size_t)(g0 + token) * D_MODEL + col] =
        (Y[token][col] - mu) * rs * gamma[col] + beta[col];
  }
}

// ---------------------------------------------------------------------------
extern "C" void kernel_launch(void* const* d_in, const int* in_sizes, int n_in,
                              void* d_out, int out_size, void* d_ws, size_t ws_size,
                              hipStream_t stream) {
  (void)in_sizes; (void)n_in; (void)out_size; (void)ws_size;
  const float* X     = (const float*)d_in[0];
  const float* adj   = (const float*)d_in[1];
  const float* Wq    = (const float*)d_in[2];
  const float* bq    = (const float*)d_in[3];
  const float* Wk    = (const float*)d_in[4];
  const float* bk    = (const float*)d_in[5];
  const float* Wv    = (const float*)d_in[6];
  const float* bv    = (const float*)d_in[7];
  const float* We    = (const float*)d_in[8];
  const float* be    = (const float*)d_in[9];
  const float* Wo    = (const float*)d_in[10];
  const float* bo    = (const float*)d_in[11];
  const float* gamma = (const float*)d_in[12];
  const float* beta  = (const float*)d_in[13];

  char* ws = (char*)d_ws;
  unsigned*       PW = (unsigned*)ws;                                // 512 KB packed weights
  unsigned short* Qp = (unsigned short*)(ws + 524288);               // 4 MB
  unsigned short* Kp = (unsigned short*)(ws + 524288 + 4194304);     // 4 MB
  unsigned*       VP = (unsigned*)(ws + 524288 + 2 * 4194304);       // 4 MB (swizzled)
  unsigned short* AO = (unsigned short*)(ws + 524288 + 3 * 4194304); // 4 MB

  pack_w   <<<512, 256, 0, stream>>>(Wq, Wk, Wv, Wo, PW);
  gat_qkv  <<<512, 256, 0, stream>>>(X, bq, bk, bv, PW, Qp, Kp, VP);
  gat_attn <<<512, 128, 0, stream>>>(adj, We, be,
                                     (const unsigned*)Qp, (const unsigned*)Kp,
                                     VP, AO);
  gat_out_ln<<<512, 256, 0, stream>>>(X, (const unsigned*)AO, PW + 3 * 32768,
                                      bo, gamma, beta, (float*)d_out);
}
